// GRU_23227183137444
// MI455X (gfx1250) — matrix operand order repaired and checked by reading, compiled, not run on hardware
//
#include <hip/hip_runtime.h>
#include <hip/hip_bf16.h>
#include <math.h>

// ---------------------------------------------------------------------------
// GRU on MI455X (gfx1250): batch-partitioned persistent scan with f16 WMMA.
//   T=512, B=256, D=H=512, O=256.
// Each workgroup owns BM=16 batch rows; h state lives in LDS for all 512
// steps. Weights are repacked once (f32 -> f16, WMMA-B-tile layout) into ws.
// gemm4/gemm2: one A-tile LDS load feeds 4/2 independent WMMA accumulators
// (z+r gates x 2 N-tiles) for A-reuse and WMMA ILP.
// ---------------------------------------------------------------------------

typedef _Float16 half_t;
typedef __attribute__((ext_vector_type(16))) _Float16 v16h;
typedef __attribute__((ext_vector_type(8)))  float    v8f;
typedef __attribute__((ext_vector_type(4)))  int      v4i;

#define T_STEPS 512
#define BATCH   256
#define DD      512   // input size (== K of all GEMMs)
#define HH      512   // hidden size
#define OO      256   // output size
#define BM      16    // batch rows per workgroup
#define SR      520   // LDS row stride in halves (1040B: 16B aligned, bank-spread)
#define MSZ     (512*512)   // elements per packed square weight matrix

union V16U { v4i i[2]; v16h h; };

// --- A operand: 16xK tile from LDS, row-major (stride SR halves). ----------
// A layout (contiguous K, matches packed B): lane<16 -> M=lane, K = k0+{0..15};
// lane>=16 -> M=lane-16, K offset +16.
__device__ __forceinline__ v16h load_a_lds(const half_t* base, int k0, int lane) {
  const int row  = lane & 15;
  const int koff = (lane < 16) ? 0 : 16;
  const half_t* p = base + row * SR + k0 + koff;
  V16U u;
  u.i[0] = *reinterpret_cast<const v4i*>(p);        // K = k0+koff   .. +7
  u.i[1] = *reinterpret_cast<const v4i*>(p + 8);    // K = k0+koff+8 .. +15
  return u.h;
}

// --- B operand: packed tiles in global memory. -----------------------------
// Tile (kc,nc) of a KxN matrix stored as 32 lanes x 16 halves contiguous:
//   elem j of lane L = src[(kc*32 + (L>>4)*16 + j)*N + nc*16 + (L&15)]
__device__ __forceinline__ v16h load_b_packed(const half_t* pk, int kc, int nc,
                                              int NT, int lane) {
  const half_t* p = pk + ((size_t)((kc * NT) + nc) * 32 + lane) * 16;
  V16U u;
  u.i[0] = *reinterpret_cast<const v4i*>(p);
  u.i[1] = *reinterpret_cast<const v4i*>(p + 8);
  return u.h;
}

__device__ __forceinline__ v8f wmma_(v16h a, v16h b, v8f c) {
  return __builtin_amdgcn_wmma_f32_16x16x32_f16(false, a, false, b,
                                                (short)0, c, false, false);
}

// One A load -> 4 independent accumulators (2 weight matrices x 2 N-tiles).
__device__ __forceinline__ void gemm4(const half_t* a_lds,
                                      const half_t* bp0, const half_t* bp1,
                                      int nc0, int NT, int KT, int lane,
                                      v8f& c00, v8f& c01, v8f& c10, v8f& c11) {
#pragma unroll 2
  for (int kc = 0; kc < KT; ++kc) {
    v16h a   = load_a_lds(a_lds, kc * 32, lane);
    v16h b00 = load_b_packed(bp0, kc, nc0,     NT, lane);
    v16h b01 = load_b_packed(bp0, kc, nc0 + 1, NT, lane);
    v16h b10 = load_b_packed(bp1, kc, nc0,     NT, lane);
    v16h b11 = load_b_packed(bp1, kc, nc0 + 1, NT, lane);
    c00 = wmma_(a, b00, c00);
    c01 = wmma_(a, b01, c01);
    c10 = wmma_(a, b10, c10);
    c11 = wmma_(a, b11, c11);
  }
}

// One A load -> 2 independent accumulators (2 N-tiles of one weight matrix).
__device__ __forceinline__ void gemm2(const half_t* a_lds, const half_t* bpk,
                                      int nc0, int NT, int KT, int lane,
                                      v8f& c0, v8f& c1) {
#pragma unroll 4
  for (int kc = 0; kc < KT; ++kc) {
    v16h a  = load_a_lds(a_lds, kc * 32, lane);
    v16h b0 = load_b_packed(bpk, kc, nc0,     NT, lane);
    v16h b1 = load_b_packed(bpk, kc, nc0 + 1, NT, lane);
    c0 = wmma_(a, b0, c0);
    c1 = wmma_(a, b1, c1);
  }
}

// Single-tile version (final FC).
__device__ __forceinline__ v8f gemm1(const half_t* a_lds, const half_t* bpk,
                                     int nc, int NT, int KT, int lane, v8f acc) {
#pragma unroll 4
  for (int kc = 0; kc < KT; ++kc) {
    v16h a = load_a_lds(a_lds, kc * 32, lane);
    v16h b = load_b_packed(bpk, kc, nc, NT, lane);
    acc = wmma_(a, b, acc);
  }
  return acc;
}

__device__ __forceinline__ float sigmoidf_(float v) {
  return 1.0f / (1.0f + __expf(-v));
}

// --- weight repack: f32 KxN row-major -> f16 WMMA-B tile layout ------------
__global__ void pack_b_kernel(const float* __restrict__ src,
                              half_t* __restrict__ dst, int K, int N) {
  const int KT = K / 32, NT = N / 16;
  const int total = KT * NT * 32;
  int idx = blockIdx.x * blockDim.x + threadIdx.x;
  if (idx >= total) return;
  const int lane = idx & 31;
  const int tile = idx >> 5;
  const int nc = tile % NT, kc = tile / NT;
  const int koff = (lane >> 4) * 16;
  const int n = nc * 16 + (lane & 15);
  half_t* d = dst + (size_t)idx * 16;
#pragma unroll
  for (int j = 0; j < 16; ++j) {
    int k = kc * 32 + koff + j;
    d[j] = (half_t)src[(size_t)k * N + n];
  }
}

// --- persistent GRU scan ---------------------------------------------------
__global__ __launch_bounds__(512, 1)
void gru_scan_kernel(const float* __restrict__ x,
                     const half_t* __restrict__ pWz, const half_t* __restrict__ pWr,
                     const half_t* __restrict__ pWh, const half_t* __restrict__ pUz,
                     const half_t* __restrict__ pUr, const half_t* __restrict__ pUh,
                     const half_t* __restrict__ pFc,
                     const float* __restrict__ bz,  const float* __restrict__ ubz,
                     const float* __restrict__ br,  const float* __restrict__ ubr,
                     const float* __restrict__ bh,  const float* __restrict__ ubh,
                     const float* __restrict__ fcb,
                     float* __restrict__ out) {
  __shared__ __align__(16) half_t sh [BM * SR];  // h state (f16)
  __shared__ __align__(16) half_t srh[BM * SR];  // r * h
  __shared__ __align__(16) half_t sx [BM * SR];  // x_t staged as f16

  const int tid  = threadIdx.x;
  const int lane = tid & 31;
  const int wave = tid >> 5;            // 16 waves
  const int b0   = blockIdx.x * BM;     // batch rows owned by this WG

  // zero-initialize h
  for (int i = tid; i < BM * SR; i += 512) sh[i] = (half_t)0.0f;

  const int NT = HH / 16;               // 32 N-tiles across hidden dim
  const int KT = DD / 32;               // 16 K-chunks
  const int nc0 = wave * 2;                          // this wave's 2 N-tiles
  const int c0 = nc0 * 16 + (lane & 15);             // absolute column, tile 0
  const int c1 = c0 + 16;                            // tile 1
  const int mrow = (lane < 16) ? 0 : 8;              // C/D layout row base

  // hoist biases (input bias + recurrent bias fused)
  const float bzz0 = bz[c0] + ubz[c0], bzz1 = bz[c1] + ubz[c1];
  const float brr0 = br[c0] + ubr[c0], brr1 = br[c1] + ubr[c1];
  const float bhh0 = bh[c0] + ubh[c0], bhh1 = bh[c1] + ubh[c1];

  __syncthreads();

  for (int t = 0; t < T_STEPS; ++t) {
    // ---- stage x_t (f32 -> f16) into LDS, coalesced ----
    const float* xt = x + ((size_t)t * BATCH + b0) * DD;
    for (int i = tid; i < BM * DD; i += 512) {
      int m = i >> 9, d = i & (DD - 1);
      sx[m * SR + d] = (half_t)xt[m * DD + d];
    }
    __syncthreads();

    // ---- phase 1: z and r gates; x@W and h@U fused into same accumulators.
    v8f az0 = {0.f,0.f,0.f,0.f,0.f,0.f,0.f,0.f};
    v8f az1 = az0, ar0 = az0, ar1 = az0;
    gemm4(sx, pWz, pWr, nc0, NT, KT, lane, az0, az1, ar0, ar1);
    gemm4(sh, pUz, pUr, nc0, NT, KT, lane, az0, az1, ar0, ar1);

    v8f z0, z1;
#pragma unroll
    for (int v = 0; v < 8; ++v) {
      const int m = mrow + v;
      z0[v] = sigmoidf_(az0[v] + bzz0);
      z1[v] = sigmoidf_(az1[v] + bzz1);
      float r0 = sigmoidf_(ar0[v] + brr0);
      float r1 = sigmoidf_(ar1[v] + brr1);
      srh[m * SR + c0] = (half_t)(r0 * (float)sh[m * SR + c0]);
      srh[m * SR + c1] = (half_t)(r1 * (float)sh[m * SR + c1]);
    }
    __syncthreads();   // srh fully written

    // ---- phase 2: candidate h_hat and blend ----
    v8f ah0 = {0.f,0.f,0.f,0.f,0.f,0.f,0.f,0.f}, ah1 = ah0;
    gemm2(sx,  pWh, nc0, NT, KT, lane, ah0, ah1);
    gemm2(srh, pUh, nc0, NT, KT, lane, ah0, ah1);

#pragma unroll
    for (int v = 0; v < 8; ++v) {
      const int m = mrow + v;
      float hold0 = (float)sh[m * SR + c0];
      float hold1 = (float)sh[m * SR + c1];
      float hh0 = tanhf(ah0[v] + bhh0);
      float hh1 = tanhf(ah1[v] + bhh1);
      // each wave writes only its own columns -> no cross-wave conflict
      sh[m * SR + c0] = (half_t)(hold0 + z0[v] * (hh0 - hold0));
      sh[m * SR + c1] = (half_t)(hold1 + z1[v] * (hh1 - hold1));
    }
    __syncthreads();   // h ready (and sx free) for next step
  }

  // ---- final FC: out = h_T @ fc_w + fc_b ----
  const int ncf = wave;                 // 16 waves x 16 cols = 256 = OO
  const int cf  = ncf * 16 + (lane & 15);
  v8f ao = {0.f,0.f,0.f,0.f,0.f,0.f,0.f,0.f};
  ao = gemm1(sh, pFc, ncf, OO / 16, DD / 32, lane, ao);
  const float fb = fcb[cf];
#pragma unroll
  for (int v = 0; v < 8; ++v) {
    const int m = mrow + v;
    out[(size_t)(b0 + m) * OO + cf] = ao[v] + fb;
  }
}

// ---------------------------------------------------------------------------
extern "C" void kernel_launch(void* const* d_in, const int* in_sizes, int n_in,
                              void* d_out, int out_size, void* d_ws, size_t ws_size,
                              hipStream_t stream) {
  const float* x    = (const float*)d_in[0];
  const float* w_z  = (const float*)d_in[1];
  const float* b_z  = (const float*)d_in[2];
  const float* u_z  = (const float*)d_in[3];
  const float* ub_z = (const float*)d_in[4];
  const float* w_r  = (const float*)d_in[5];
  const float* b_r  = (const float*)d_in[6];
  const float* u_r  = (const float*)d_in[7];
  const float* ub_r = (const float*)d_in[8];
  const float* w_h  = (const float*)d_in[9];
  const float* b_h  = (const float*)d_in[10];
  const float* u_h  = (const float*)d_in[11];
  const float* ub_h = (const float*)d_in[12];
  const float* fc_w = (const float*)d_in[13];
  const float* fc_b = (const float*)d_in[14];
  float* out = (float*)d_out;

  half_t* ws  = (half_t*)d_ws;
  half_t* pWz = ws + 0 * (size_t)MSZ;
  half_t* pWr = ws + 1 * (size_t)MSZ;
  half_t* pWh = ws + 2 * (size_t)MSZ;
  half_t* pUz = ws + 3 * (size_t)MSZ;
  half_t* pUr = ws + 4 * (size_t)MSZ;
  half_t* pUh = ws + 5 * (size_t)MSZ;
  half_t* pFc = ws + 6 * (size_t)MSZ;   // 512x256

  // repack weights (f32 -> f16 WMMA-tile layout), every call (deterministic)
  const int sq_threads = (512 / 32) * (512 / 16) * 32;   // 16384
  const int fc_threads = (512 / 32) * (256 / 16) * 32;   // 8192
  dim3 blk(256);
  pack_b_kernel<<<dim3((sq_threads + 255) / 256), blk, 0, stream>>>(w_z, pWz, 512, 512);
  pack_b_kernel<<<dim3((sq_threads + 255) / 256), blk, 0, stream>>>(w_r, pWr, 512, 512);
  pack_b_kernel<<<dim3((sq_threads + 255) / 256), blk, 0, stream>>>(w_h, pWh, 512, 512);
  pack_b_kernel<<<dim3((sq_threads + 255) / 256), blk, 0, stream>>>(u_z, pUz, 512, 512);
  pack_b_kernel<<<dim3((sq_threads + 255) / 256), blk, 0, stream>>>(u_r, pUr, 512, 512);
  pack_b_kernel<<<dim3((sq_threads + 255) / 256), blk, 0, stream>>>(u_h, pUh, 512, 512);
  pack_b_kernel<<<dim3((fc_threads + 255) / 256), blk, 0, stream>>>(fc_w, pFc, 512, 256);

  // persistent scan: one workgroup per 16 batch rows, 512 threads (16 waves)
  gru_scan_kernel<<<dim3(BATCH / BM), dim3(512), 0, stream>>>(
      x, pWz, pWr, pWh, pUz, pUr, pUh, pFc,
      b_z, ub_z, b_r, ub_r, b_h, ub_h, fc_b, out);
}
